// TconvLayer_18356690223735
// MI455X (gfx1250) — matrix operand order repaired and checked by reading, not compiled, not run on hardware
//
#include <hip/hip_runtime.h>

#define B_   8
#define CIN  128
#define HW   64
#define COUT 128
#define MID  16
#define NPIX (B_ * HW * HW)

#define ROWS   6        // slab rows: h0-2 .. h0+3 (covers dil=1 and dil=2)
#define WSLOTS 68       // slab cols: w-2 .. w+65  (padded NHWC buffer width)
#define HP     68       // padded spatial extent in xbfp
#define LDC    136      // padded bf16 c-stride in LDS (TDM pad: +4DW / 64DW)
#define SLABW  (WSLOTS * CIN)   // 8704 contiguous bf16 per padded row

typedef __attribute__((ext_vector_type(16))) __bf16          bf16x16;
typedef __attribute__((ext_vector_type(8)))  float           f32x8;
typedef __attribute__((ext_vector_type(8)))  unsigned short  u16x8;
typedef __attribute__((ext_vector_type(16))) unsigned short  u16x16;
typedef __attribute__((ext_vector_type(4)))  unsigned        u32x4;
typedef __attribute__((ext_vector_type(8)))  unsigned        u32x8;

// fp32 -> bf16 round-to-nearest-even
__device__ __forceinline__ unsigned short f2bf(float f) {
    unsigned u = __builtin_bit_cast(unsigned, f);
    u += 0x7FFFu + ((u >> 16) & 1u);
    return (unsigned short)(u >> 16);
}

// 16-element bf16 fragment from two contiguous 8-element chunks (LDS or global)
__device__ __forceinline__ bf16x16 frag16(const unsigned short* p0,
                                          const unsigned short* p1) {
    u16x8 lo = *(const u16x8*)p0;
    u16x8 hi = *(const u16x8*)p1;
    u16x16 c = __builtin_shufflevector(lo, hi, 0, 1, 2, 3, 4, 5, 6, 7,
                                               8, 9, 10, 11, 12, 13, 14, 15);
    return __builtin_bit_cast(bf16x16, c);
}

// ---------------------------------------------------------------------------
// Kernel W: weight transpose + fp32->bf16.
//   wbf[((tap*COUT + r)*MID + m)*CIN + c]  (c-contiguous == A-fragment order)
// ---------------------------------------------------------------------------
__global__ __launch_bounds__(256)
void wprep_kernel(const float* __restrict__ w_mid,
                  unsigned short* __restrict__ wbf) {
    const int o   = blockIdx.x * 256 + threadIdx.x;
    const int c   = o & 127;
    const int m   = (o >> 7) & 15;
    const int r   = (o >> 11) & 127;
    const int tap = o >> 18;
    wbf[o] = f2bf(w_mid[(((r * MID + m) * CIN + c) * 9) + tap]);
}

// ---------------------------------------------------------------------------
// Kernel X: x -> zero-padded NHWC bf16:  xbfp[b][hp][wp][c], hp/wp in [0,68),
// h = hp-2, w = wp-2 (halo rows/cols are zeros).  LDS-tiled transpose:
// coalesced fp32 reads (w-contiguous), coalesced bf16 writes (c-contiguous).
// One block per (b, hp) row.
// ---------------------------------------------------------------------------
__global__ __launch_bounds__(256)
void xprep_kernel(const float* __restrict__ x,
                  unsigned short* __restrict__ xbfp) {
    __shared__ unsigned short T[CIN * (WSLOTS + 1)];   // +1 pad vs bank conflicts

    const int bh = blockIdx.x;            // 0 .. B_*HP-1
    const int b  = bh / HP;
    const int hp = bh % HP;
    const int h  = hp - 2;

    // load phase: e = c*68 + wp  (consecutive tid -> consecutive w)
#pragma unroll
    for (int k = 0; k < (CIN * WSLOTS) / 256; ++k) {   // 8704/256 = 34
        const int e  = k * 256 + threadIdx.x;
        const int c  = e / WSLOTS;
        const int wp = e % WSLOTS;
        const int w  = wp - 2;
        float v = 0.f;
        if ((unsigned)h < (unsigned)HW && (unsigned)w < (unsigned)HW)
            v = x[((b * CIN + c) * HW + h) * HW + w];
        T[c * (WSLOTS + 1) + wp] = f2bf(v);
    }
    __syncthreads();

    // store phase: e = wp*128 + c  (consecutive tid -> consecutive c)
#pragma unroll
    for (int k = 0; k < (CIN * WSLOTS) / 256; ++k) {
        const int e  = k * 256 + threadIdx.x;
        const int wp = e >> 7;
        const int c  = e & 127;
        xbfp[((bh * WSLOTS) + wp) * CIN + c] = T[c * (WSLOTS + 1) + wp];
    }
}

// ---------------------------------------------------------------------------
// Kernel 1: fused dilated conv (implicit GEMM, bf16 WMMA f32-acc) + LeakyReLU
//           + grouped 1x1 pointwise + bias; BN partials via f32 atomics.
// Grid: (B*H/2 = 256, branch-group = 16, parity = 2), block = 256 (8 waves).
// Tile: M = 64 mids (4 same-dilation branches) x N = 128 px (2 rows x 64).
// x slab is DMA'd once by the Tensor Data Mover with LDS padding so rows land
// at stride 136 bf16 (conflict-free ds_load_b128); all 9 taps served from it.
// A fragments come straight from the pre-transposed global weights (L2-hot).
// ---------------------------------------------------------------------------
__global__ __launch_bounds__(256)
void conv_wmma_kernel(const unsigned short* __restrict__ xbfp,
                      const unsigned short* __restrict__ wbf,
                      const float* __restrict__ w_pt,
                      const float* __restrict__ b_pt,
                      float* __restrict__ ybuf,
                      float* __restrict__ wsum,
                      float* __restrict__ wssq) {
    __shared__ unsigned short Xlds[ROWS * WSLOTS * LDC];   // ~108 KB padded slab
    __shared__ float wptl[64];
    __shared__ float bptl[4];

    const int tid    = threadIdx.x;
    const int parity = blockIdx.z;              // 0 -> dil 1, 1 -> dil 2
    const int dil    = 1 + parity;
    const int g      = blockIdx.y;              // branch-group (4 branches)
    const int b      = blockIdx.x >> 5;         // image
    const int h0     = (blockIdx.x & 31) << 1;  // rows h0, h0+1

    if (tid < 64) {
        const int jb = tid >> 4;
        const int r  = (((g << 2) + jb) << 1) + parity;
        wptl[tid] = w_pt[r * MID + (tid & 15)];
        if ((tid & 15) == 0) bptl[jb] = b_pt[r];
    }

    // ---- TDM: DMA 6 x 8704 bf16 (rows hp0..hp0+5) global -> LDS, padded ----
    if (tid < 32) {   // one wave issues the tensor op (EXEC-independent DMA)
        const unsigned lds_base = (unsigned)(size_t)(void*)&Xlds[0];
        const unsigned long long ga =
            (unsigned long long)(const void*)(xbfp + (size_t)(b * HP + h0) * SLABW);

        // D# group 0: count=1 | lds_addr | global_addr[56:0] | type=2
        const u32x4 g0 = {1u,
                          lds_base,
                          (unsigned)(ga & 0xFFFFFFFFull),
                          (unsigned)((ga >> 32) & 0x1FFFFFFull) | 0x80000000u};
        // D# group 1: data_size=1(2B) pad_enable pad_interval=5(64DW)
        //             pad_amount=3(4DW); tensor_dim0=tile_dim0=8704;
        //             tensor_dim1=tile_dim1=6; dim0_stride=8704
        const u32x8 g1 = {0x07510000u,            // sizes/pad flags
                          (unsigned)(SLABW << 16),// tensor_dim0[15:0] in [31:16]
                          (unsigned)(ROWS << 16), // tensor_dim1[15:0] in [31:16]
                          (unsigned)(SLABW << 16) | 0u, // tile_dim0 in [31:16]
                          (unsigned)ROWS,         // tile_dim1
                          (unsigned)SLABW,        // tensor_dim0_stride[31:0]
                          0u, 0u};
        asm volatile("tensor_load_to_lds %0, %1"
                     :
                     : "s"(g0), "s"(g1)
                     : "memory");
        __builtin_amdgcn_s_wait_tensorcnt(0);
    }
    __syncthreads();    // slab + wptl visible to all waves

    const int lane = tid & 31;
    const int wid  = tid >> 5;
    const int wm = wid >> 2, wn = wid & 3;      // 2 x 4 wave grid
    const int no = wn << 5;
    const int hb = lane >> 4, ln = lane & 15;

    // lane-constant fragment bases
    const unsigned short* abase[2];
    int bbase[2];
#pragma unroll
    for (int mi = 0; mi < 2; ++mi) {
        const int r = (((g << 2) + (wm << 1) + mi) << 1) + parity;
        abase[mi] = wbf + (r * MID + ln) * CIN + (hb << 4);
    }
#pragma unroll
    for (int ni = 0; ni < 2; ++ni) {
        const int px = no + (ni << 4) + ln;
        const int rr = px >> 6, w = px & 63;
        bbase[ni] = (rr * WSLOTS + w) * LDC + (hb << 4);
    }

    f32x8 acc[2][2];
#pragma unroll
    for (int mi = 0; mi < 2; ++mi)
#pragma unroll
        for (int ni = 0; ni < 2; ++ni)
            acc[mi][ni] = (f32x8){0.f, 0.f, 0.f, 0.f, 0.f, 0.f, 0.f, 0.f};

#pragma unroll
    for (int tap = 0; tap < 9; ++tap) {
        const int kh = tap / 3, kw = tap % 3;                 // constants
        const int tapofs = ((dil * (kh - 1) + 2) * WSLOTS +
                            (dil * (kw - 1) + 2)) * LDC;      // scalar per tap
        const int wofs = tap * (COUT * MID * CIN);
#pragma unroll
        for (int kc = 0; kc < 4; ++kc) {
            const int c0 = kc << 5;
            bf16x16 afr[2], bfr[2];
#pragma unroll
            for (int mi = 0; mi < 2; ++mi) {   // A elem j -> K = 16*hb + j
                const unsigned short* p = abase[mi] + wofs + c0;
                afr[mi] = frag16(p, p + 8);
            }
#pragma unroll
            for (int ni = 0; ni < 2; ++ni) {   // B elem j -> K = 16*hb + j
                const unsigned short* p = &Xlds[bbase[ni] + tapofs + c0];
                bfr[ni] = frag16(p, p + 8);
            }
#pragma unroll
            for (int mi = 0; mi < 2; ++mi)
#pragma unroll
                for (int ni = 0; ni < 2; ++ni)
                    acc[mi][ni] = __builtin_amdgcn_wmma_f32_16x16x32_bf16(
                        false, afr[mi], false, bfr[ni],
                        (short)0, acc[mi][ni], false, false);
        }
    }

    // ---- epilogue: LeakyReLU + pointwise(16) + bias; y store + BN partials ----
#pragma unroll
    for (int mi = 0; mi < 2; ++mi) {
        const int jb = (wm << 1) + mi;              // 16-row WMMA tile == 1 branch
        const int r  = (((g << 2) + jb) << 1) + parity;
        float s1 = 0.f, s2 = 0.f;
#pragma unroll
        for (int ni = 0; ni < 2; ++ni) {
            float p = 0.f;
#pragma unroll
            for (int v = 0; v < 8; ++v) {           // C/D elem v -> M = v + 8*hb
                float a = acc[mi][ni][v];
                a = (a >= 0.f) ? a : 0.1f * a;      // LeakyReLU(0.1)
                p += wptl[(jb << 4) + v + (hb << 3)] * a;
            }
            p += __shfl_xor(p, 16, 32);             // combine M halves
            const float yv = p + bptl[jb];
            if (hb == 0) {
                const int px = no + (ni << 4) + ln;
                const int rr = px >> 6, w = px & 63;
                ybuf[((b * COUT + r) * HW + (h0 + rr)) * HW + w] = yv;
                s1 += yv;
                s2 += yv * yv;
            }
        }
#pragma unroll
        for (int off = 16; off > 0; off >>= 1) {
            s1 += __shfl_xor(s1, off, 32);
            s2 += __shfl_xor(s2, off, 32);
        }
        if (lane == 0) {
            atomicAdd(&wsum[r], s1);
            atomicAdd(&wssq[r], s2);
        }
    }
}

// ---------------------------------------------------------------------------
// Kernel 0: zero the stat accumulators each launch
// ---------------------------------------------------------------------------
__global__ void zero_stats_kernel(float* __restrict__ p) {
    p[threadIdx.x] = 0.f;
}

// ---------------------------------------------------------------------------
// Kernel 2: BatchNorm (batch stats, biased var) + ReLU, float4 traffic.
// ---------------------------------------------------------------------------
__global__ __launch_bounds__(256)
void bn_relu_kernel(const float* __restrict__ ybuf,
                    const float* __restrict__ wsum,
                    const float* __restrict__ wssq,
                    const float* __restrict__ gamma,
                    const float* __restrict__ beta,
                    float* __restrict__ out) {
    const int r = blockIdx.x;
    const int b = blockIdx.y;
    const float invN  = 1.f / (float)NPIX;
    const float mean  = wsum[r] * invN;
    const float var   = wssq[r] * invN - mean * mean;
    const float scale = gamma[r] * rsqrtf(var + 1e-5f);
    const float shift = beta[r] - mean * scale;

    const int base = (b * COUT + r) * HW * HW;
    const float4* src = (const float4*)(ybuf + base);
    float4*       dst = (float4*)(out + base);
#pragma unroll 4
    for (int i = threadIdx.x; i < (HW * HW) / 4; i += 256) {
        float4 v = src[i];
        v.x = fmaxf(0.f, v.x * scale + shift);
        v.y = fmaxf(0.f, v.y * scale + shift);
        v.z = fmaxf(0.f, v.z * scale + shift);
        v.w = fmaxf(0.f, v.w * scale + shift);
        dst[i] = v;
    }
}

extern "C" void kernel_launch(void* const* d_in, const int* in_sizes, int n_in,
                              void* d_out, int out_size, void* d_ws, size_t ws_size,
                              hipStream_t stream) {
    const float* x     = (const float*)d_in[0];
    const float* w_mid = (const float*)d_in[1];
    const float* w_pt  = (const float*)d_in[2];
    const float* b_pt  = (const float*)d_in[3];
    const float* gamma = (const float*)d_in[4];
    const float* beta  = (const float*)d_in[5];
    float* out = (float*)d_out;

    float* ws   = (float*)d_ws;
    float* wsum = ws;                          // [128]
    float* wssq = ws + 128;                    // [128]
    float* ybuf = ws + 256;                    // [B,COUT,H,W] pre-BN (16.8 MB)
    unsigned short* wbf  = (unsigned short*)(ws + 256 + (size_t)B_ * COUT * HW * HW);
                                               // [9][COUT][MID][CIN] bf16 (4.7 MB)
    unsigned short* xbfp = wbf + (size_t)9 * COUT * MID * CIN;
                                               // [B][68][68][CIN] bf16 (9.5 MB)

    zero_stats_kernel<<<1, 256, 0, stream>>>(ws);

    const int wtot = 9 * COUT * MID * CIN;     // 2,359,296
    wprep_kernel<<<wtot / 256, 256, 0, stream>>>(w_mid, wbf);

    xprep_kernel<<<B_ * HP, 256, 0, stream>>>(x, xbfp);

    dim3 grid1(B_ * (HW / 2), 16, 2);
    conv_wmma_kernel<<<grid1, 256, 0, stream>>>(xbfp, wbf, w_pt, b_pt,
                                                ybuf, wsum, wssq);

    dim3 grid2(COUT, B_);
    bn_relu_kernel<<<grid2, 256, 0, stream>>>(ybuf, wsum, wssq, gamma, beta, out);
}
